// FractionalAttention_54786602828151
// MI455X (gfx1250) — matrix operand order repaired and checked by reading, not compiled, not run on hardware
//
#include <hip/hip_runtime.h>
#include <math.h>

// ---------------------------------------------------------------------------
// FractionalAttention for MI455X (gfx1250, wave32, WMMA).
// All matmuls use v_wmma_f32_16x16x32_bf16 (bf16 A/B, fp32 accumulate).
// V and attention-context are stored TRANSPOSED ([b,h,dk,S]) so every WMMA
// B-operand in the pipeline is a contiguous 32-byte load (global_load_b128).
// Pipeline:
//   1) cvt x, Wq, Wk, Wv, Wo -> bf16 workspace
//   2) GL coefficient vector w[0..S-1] (scaled by h^{-alpha})
//   3) Q/K/V projection GEMMs (WMMA, 16x32 C-tile/wave); V written transposed
//   4) flash attention per 16-row Q tile (WMMA QK^T, online softmax, WMMA PV,
//      prefetch of next K/V chunk); context written transposed
//   5) causal Toeplitz matmul: A-tile generated in-register from w[], WMMA
//   6) output projection GEMM (WMMA) -> fp32
//   7) residual + LayerNorm epilogue -> d_out (fp32)
// ---------------------------------------------------------------------------

typedef __attribute__((ext_vector_type(16))) __bf16 v16bf;
typedef __attribute__((ext_vector_type(8)))  __bf16 v8bf;
typedef __attribute__((ext_vector_type(8)))  float  v8f;

#define SEQ   2048
#define DIM   1024
#define NH    16
#define DK    64
#define NB    2
#define NROW  (NB * SEQ)       // 4096
#define ALPHA_F 0.5f
#define LN_EPSF 1e-5f

__device__ __forceinline__ v8f wmma_bf16(v16bf a, v16bf b, v8f c) {
  // (neg_a, A, neg_b, B, c_mod, C, reuse_a, reuse_b)
  return __builtin_amdgcn_wmma_f32_16x16x32_bf16(false, a, false, b,
                                                 (short)0, c, false, false);
}

// A operand (16x32 bf16) from row-major [M,K]: lane holds row lane%16;
// element e -> K = (e&7) + (e>=8 ? 16 : 0) + (lane/16)*8  (ISA 7.12.2).
__device__ __forceinline__ v16bf load_a_rm(const __bf16* base, int ldk,
                                           int row, int k0, int half) {
  const __bf16* p = base + (size_t)row * ldk + k0 + half * 8;
  v8bf lo = *(const v8bf*)p;          // K = k0+half*8 .. +7
  v8bf hi = *(const v8bf*)(p + 16);   // K = k0+half*8+16 .. +23
  v16bf a;
#pragma unroll
  for (int i = 0; i < 8; ++i) { a[i] = lo[i]; a[i + 8] = hi[i]; }
  return a;
}

// B operand (32x16) from [N,K] row-major memory (mem[n][k] = B[k][n]):
// lane -> n = n0+lane%16; element e -> K = k0 + (e&7) + (e>=8 ? 16 : 0) +
// (lane/16)*8 (same order as A). Used for weights, K-matrix, V^T and ctx^T.
__device__ __forceinline__ v16bf load_b_nk(const __bf16* base, int ldk,
                                           int n0, int k0, int lane) {
  int n = n0 + (lane & 15), half = lane >> 4;
  const __bf16* p = base + (size_t)n * ldk + k0 + half * 8;
  v8bf lo = *(const v8bf*)p;
  v8bf hi = *(const v8bf*)(p + 16);
  v16bf b;
#pragma unroll
  for (int i = 0; i < 8; ++i) { b[i] = lo[i]; b[i + 8] = hi[i]; }
  return b;
}

// Reductions across the 16 lanes sharing one C-matrix row (wave32 halves).
__device__ __forceinline__ float redmax16(float v) {
#pragma unroll
  for (int m = 1; m < 16; m <<= 1) v = fmaxf(v, __shfl_xor(v, m, 32));
  return v;
}
__device__ __forceinline__ float redsum16(float v) {
#pragma unroll
  for (int m = 1; m < 16; m <<= 1) v += __shfl_xor(v, m, 32);
  return v;
}

// ---------------------------------------------------------------------------
__global__ __launch_bounds__(256) void cvt_bf16_kernel(
    const float* __restrict__ src, __bf16* __restrict__ dst, int n) {
  int i = blockIdx.x * 256 + threadIdx.x;
  if (i < n) dst[i] = (__bf16)src[i];
}

// Grunwald-Letnikov coefficients, pre-scaled by h^{-alpha} = sqrt(S-1).
__global__ void glcoef_kernel(float* __restrict__ wc) {
  if (threadIdx.x == 0 && blockIdx.x == 0) {
    float scale = sqrtf((float)(SEQ - 1));
    float w = 1.0f;
    wc[0] = scale;
    for (int j = 1; j < SEQ; ++j) {
      w *= ((float)j - 1.0f - ALPHA_F) / (float)j;
      wc[j] = scale * w;
    }
  }
}

// ---------------------------------------------------------------------------
// Q/K/V projections: C[m,n] = sum_k xb[m,k] * W[n,k] + bias[n]
// One 16x32 C tile per wave (A reused across 2 WMMAs per k-step).
// blockIdx.y selects q/k/v. Q,K stored [b,h,s,d]; V stored transposed
// [b,h,d,s] so the PV B-operand is contiguous.
__global__ __launch_bounds__(128) void proj_gemm_kernel(
    const __bf16* __restrict__ xb,
    const __bf16* __restrict__ w0, const __bf16* __restrict__ w1,
    const __bf16* __restrict__ w2,
    const float* __restrict__ b0, const float* __restrict__ b1,
    const float* __restrict__ b2,
    __bf16* __restrict__ q, __bf16* __restrict__ k, __bf16* __restrict__ v) {
  const __bf16* W; const float* bias; __bf16* out;
  int z = blockIdx.y;
  switch (z) {
    case 0:  W = w0; bias = b0; out = q; break;
    case 1:  W = w1; bias = b1; out = k; break;
    default: W = w2; bias = b2; out = v; break;
  }
  int lane = threadIdx.x & 31, wv = threadIdx.x >> 5;
  int wid = blockIdx.x * 4 + wv;
  int ntile = wid & 31, mtile = wid >> 5;       // 32 n-tiles(x32), 256 m-tiles
  int m0 = mtile * 16, n0 = ntile * 32;
  int lm = lane & 15, half = lane >> 4;
  v8f acc0 = {}, acc1 = {};
  for (int k0 = 0; k0 < DIM; k0 += 32) {
    v16bf a   = load_a_rm(xb, DIM, m0 + lm, k0, half);
    v16bf bt0 = load_b_nk(W, DIM, n0,      k0, lane);
    v16bf bt1 = load_b_nk(W, DIM, n0 + 16, k0, lane);
    acc0 = wmma_bf16(a, bt0, acc0);
    acc1 = wmma_bf16(a, bt1, acc1);
  }
#pragma unroll
  for (int sub = 0; sub < 2; ++sub) {
    v8f acc = sub ? acc1 : acc0;
    int n = n0 + sub * 16 + lm;
    int hh = n >> 6, d = n & 63;
    float bi = bias[n];
#pragma unroll
    for (int r = 0; r < 8; ++r) {
      int m = m0 + r + half * 8;
      int b = m >> 11, s = m & (SEQ - 1);
      float val = acc[r] + bi;
      if (z == 2)  // V transposed: [b,h,d,s]
        out[((size_t)(b * NH + hh) * DK + d) * SEQ + s] = (__bf16)val;
      else         // Q,K: [b,h,s,d]
        out[((size_t)(b * NH + hh) * SEQ + s) * DK + d] = (__bf16)val;
    }
  }
}

// ---------------------------------------------------------------------------
// Flash attention: each wave owns one (b,h, 16-row Q tile); streams K/V in
// 32-column chunks with online softmax; 16x64 fp32 output accumulator.
// V is [b,h,d,s] (transposed) -> PV B-operand loads are contiguous b128.
// Context is written transposed [b,h,d,s] for the Toeplitz stage.
__global__ __launch_bounds__(128) void flash_attn_kernel(
    const __bf16* __restrict__ qb, const __bf16* __restrict__ kb,
    const __bf16* __restrict__ vtb, __bf16* __restrict__ ctxt) {
  __shared__ __align__(32) __bf16 Plds[4][16][32];  // C->A relayout of P
  int lane = threadIdx.x & 31, wv = threadIdx.x >> 5;
  int wid = blockIdx.x * 4 + wv;
  int si = wid & (SEQ / 16 - 1);
  int bh = wid >> 7;                       // SEQ/16 == 128
  const __bf16* Q  = qb  + (size_t)bh * SEQ * DK;
  const __bf16* K  = kb  + (size_t)bh * SEQ * DK;
  const __bf16* Vt = vtb + (size_t)bh * DK * SEQ;  // [d][s]
  int lm = lane & 15, half = lane >> 4;
  v16bf aQ0 = load_a_rm(Q, DK, si * 16 + lm, 0,  half);
  v16bf aQ1 = load_a_rm(Q, DK, si * 16 + lm, 32, half);
  v8f o0 = {}, o1 = {}, o2 = {}, o3 = {};
  float mrun[8], lrun[8];
#pragma unroll
  for (int r = 0; r < 8; ++r) { mrun[r] = -1e30f; lrun[r] = 0.0f; }
  const float isc = 0.125f;                // 1/sqrt(dk)
  for (int j = 0; j < SEQ / 32; ++j) {
    int t0 = j * 32;
    if (j + 1 < SEQ / 32) {                // prefetch next K/V chunk
      int tn = t0 + 32;
      __builtin_prefetch(K + (size_t)tn * DK + lane * 128, 0, 1);       // 4KB
      __builtin_prefetch(Vt + (size_t)(2 * lane) * SEQ + tn, 0, 1);
      __builtin_prefetch(Vt + (size_t)(2 * lane + 1) * SEQ + tn, 0, 1);
    }
    v8f s0 = {}, s1 = {};
    s0 = wmma_bf16(aQ0, load_b_nk(K, DK, t0,       0, lane), s0);
    s0 = wmma_bf16(aQ1, load_b_nk(K, DK, t0,      32, lane), s0);
    s1 = wmma_bf16(aQ0, load_b_nk(K, DK, t0 + 16,  0, lane), s1);
    s1 = wmma_bf16(aQ1, load_b_nk(K, DK, t0 + 16, 32, lane), s1);
    float corr[8];
#pragma unroll
    for (int r = 0; r < 8; ++r) {
      float a = s0[r] * isc, b2 = s1[r] * isc;
      float mx = redmax16(fmaxf(a, b2));
      float mnew = fmaxf(mrun[r], mx);
      float pa = __expf(a - mnew), pb = __expf(b2 - mnew);
      float rs = redsum16(pa + pb);
      corr[r] = __expf(mrun[r] - mnew);
      lrun[r] = lrun[r] * corr[r] + rs;
      mrun[r] = mnew;
      s0[r] = pa; s1[r] = pb;
    }
#pragma unroll
    for (int r = 0; r < 8; ++r) {
      o0[r] *= corr[r]; o1[r] *= corr[r]; o2[r] *= corr[r]; o3[r] *= corr[r];
    }
    __syncthreads();                       // uniform trip count across block
#pragma unroll
    for (int r = 0; r < 8; ++r) {          // C layout -> LDS row-major P
      Plds[wv][r + half * 8][lm]      = (__bf16)s0[r];
      Plds[wv][r + half * 8][16 + lm] = (__bf16)s1[r];
    }
    __syncthreads();
    v16bf aP = load_a_rm(&Plds[wv][0][0], 32, lm, 0, half);
    // B[k=t][n=d] = Vt[d][t0+t] -> contiguous loads, ldk = SEQ
    o0 = wmma_bf16(aP, load_b_nk(Vt, SEQ,  0, t0, lane), o0);
    o1 = wmma_bf16(aP, load_b_nk(Vt, SEQ, 16, t0, lane), o1);
    o2 = wmma_bf16(aP, load_b_nk(Vt, SEQ, 32, t0, lane), o2);
    o3 = wmma_bf16(aP, load_b_nk(Vt, SEQ, 48, t0, lane), o3);
  }
  // store context transposed: ctxt[bh][d][s]
  __bf16* Ct = ctxt + (size_t)bh * DK * SEQ;
#pragma unroll
  for (int r = 0; r < 8; ++r) {
    float inv = 1.0f / lrun[r];
    int s = si * 16 + half * 8 + r;
    Ct[(size_t)(lm)      * SEQ + s] = (__bf16)(o0[r] * inv);
    Ct[(size_t)(16 + lm) * SEQ + s] = (__bf16)(o1[r] * inv);
    Ct[(size_t)(32 + lm) * SEQ + s] = (__bf16)(o2[r] * inv);
    Ct[(size_t)(48 + lm) * SEQ + s] = (__bf16)(o3[r] * inv);
  }
}

// ---------------------------------------------------------------------------
// Causal GL-Toeplitz matmul: frac[s,d] = sum_{t<=s} w[s-t] * ctx[t,d].
// A-tile generated in registers from w[]; causal K-loop stops at the diagonal.
// ctx is transposed [b,h,d,s] -> B-operand loads contiguous.
// Output bf16 [B*S, D] with heads folded back: col = h*64 + d.
__global__ __launch_bounds__(128) void frac_toeplitz_kernel(
    const __bf16* __restrict__ ctxt, const float* __restrict__ wc,
    __bf16* __restrict__ fracb) {
  int lane = threadIdx.x & 31, wv = threadIdx.x >> 5;
  int wid = blockIdx.x * 4 + wv;
  int si = wid & (SEQ / 16 - 1);
  int bh = wid >> 7;
  int b = bh >> 4, h = bh & (NH - 1);
  const __bf16* Ct = ctxt + (size_t)bh * DK * SEQ;  // [d][s]
  int lm = lane & 15, half = lane >> 4;
  int srow = si * 16 + lm;
  v8f o0 = {}, o1 = {}, o2 = {}, o3 = {};
  int nchunk = (si >> 1) + 1;              // chunks of 32 up to the diagonal
  for (int j = 0; j < nchunk; ++j) {
    int t0 = j * 32;
    v16bf aT;
#pragma unroll
    for (int e = 0; e < 16; ++e) {
      int kk = (e & 7) + ((e >> 3) << 4) + half * 8;  // A-layout K index
      int t = t0 + kk;
      aT[e] = (srow >= t) ? (__bf16)wc[srow - t] : (__bf16)0.0f;
    }
    o0 = wmma_bf16(aT, load_b_nk(Ct, SEQ,  0, t0, lane), o0);
    o1 = wmma_bf16(aT, load_b_nk(Ct, SEQ, 16, t0, lane), o1);
    o2 = wmma_bf16(aT, load_b_nk(Ct, SEQ, 32, t0, lane), o2);
    o3 = wmma_bf16(aT, load_b_nk(Ct, SEQ, 48, t0, lane), o3);
  }
  size_t rowbase = (size_t)b * SEQ + si * 16 + half * 8;
#pragma unroll
  for (int r = 0; r < 8; ++r) {
    __bf16* row = fracb + (rowbase + r) * DIM + h * DK;
    row[lm]      = (__bf16)o0[r];
    row[16 + lm] = (__bf16)o1[r];
    row[32 + lm] = (__bf16)o2[r];
    row[48 + lm] = (__bf16)o3[r];
  }
}

// ---------------------------------------------------------------------------
// Output projection: out[m,n] = sum_k frac[m,k]*Wo[n,k] + bo[n]  (fp32 out)
// 16x32 C tile per wave.
__global__ __launch_bounds__(128) void out_gemm_kernel(
    const __bf16* __restrict__ fracb, const __bf16* __restrict__ wob,
    const float* __restrict__ bo, float* __restrict__ outf) {
  int lane = threadIdx.x & 31, wv = threadIdx.x >> 5;
  int wid = blockIdx.x * 4 + wv;
  int ntile = wid & 31, mtile = wid >> 5;
  int m0 = mtile * 16, n0 = ntile * 32;
  int lm = lane & 15, half = lane >> 4;
  v8f acc0 = {}, acc1 = {};
  for (int k0 = 0; k0 < DIM; k0 += 32) {
    v16bf a   = load_a_rm(fracb, DIM, m0 + lm, k0, half);
    v16bf bt0 = load_b_nk(wob, DIM, n0,      k0, lane);
    v16bf bt1 = load_b_nk(wob, DIM, n0 + 16, k0, lane);
    acc0 = wmma_bf16(a, bt0, acc0);
    acc1 = wmma_bf16(a, bt1, acc1);
  }
#pragma unroll
  for (int sub = 0; sub < 2; ++sub) {
    v8f acc = sub ? acc1 : acc0;
    int n = n0 + sub * 16 + lm;
    float bi = bo[n];
#pragma unroll
    for (int r = 0; r < 8; ++r) {
      int m = m0 + r + half * 8;
      outf[(size_t)m * DIM + n] = acc[r] + bi;
    }
  }
}

// ---------------------------------------------------------------------------
// y = LayerNorm(x + out) * g + b, one block per row of 1024.
__global__ __launch_bounds__(256) void resid_ln_kernel(
    const float* __restrict__ x, const float* __restrict__ outf,
    const float* __restrict__ g, const float* __restrict__ be,
    float* __restrict__ y) {
  __shared__ float red[256];
  int row = blockIdx.x;
  const float* xr   = x    + (size_t)row * DIM;
  const float* orow = outf + (size_t)row * DIM;
  float v[4]; float s = 0.f, s2 = 0.f;
#pragma unroll
  for (int i = 0; i < 4; ++i) {
    int idx = threadIdx.x + i * 256;
    v[i] = xr[idx] + orow[idx];
    s += v[i]; s2 += v[i] * v[i];
  }
  red[threadIdx.x] = s; __syncthreads();
  for (int off = 128; off > 0; off >>= 1) {
    if (threadIdx.x < off) red[threadIdx.x] += red[threadIdx.x + off];
    __syncthreads();
  }
  float mean = red[0] / (float)DIM;
  __syncthreads();
  red[threadIdx.x] = s2; __syncthreads();
  for (int off = 128; off > 0; off >>= 1) {
    if (threadIdx.x < off) red[threadIdx.x] += red[threadIdx.x + off];
    __syncthreads();
  }
  float var  = red[0] / (float)DIM - mean * mean;
  float rstd = rsqrtf(var + LN_EPSF);
#pragma unroll
  for (int i = 0; i < 4; ++i) {
    int idx = threadIdx.x + i * 256;
    y[(size_t)row * DIM + idx] = (v[i] - mean) * rstd * g[idx] + be[idx];
  }
}

// ---------------------------------------------------------------------------
extern "C" void kernel_launch(void* const* d_in, const int* in_sizes, int n_in,
                              void* d_out, int out_size, void* d_ws, size_t ws_size,
                              hipStream_t stream) {
  const float* x   = (const float*)d_in[0];
  const float* Wq  = (const float*)d_in[1];
  const float* bq  = (const float*)d_in[2];
  const float* Wk  = (const float*)d_in[3];
  const float* bk  = (const float*)d_in[4];
  const float* Wv  = (const float*)d_in[5];
  const float* bv  = (const float*)d_in[6];
  const float* Wo  = (const float*)d_in[7];
  const float* bo  = (const float*)d_in[8];
  const float* lng = (const float*)d_in[9];
  const float* lnb = (const float*)d_in[10];
  float* y = (float*)d_out;

  // Workspace layout (~56 MB total). outf aliases qb/kb (dead post-attention).
  char* ws = (char*)d_ws;
  const size_t MB = 1ull << 20;
  __bf16* qb    = (__bf16*)(ws + 0  * MB);  // [B,H,S,dk] bf16, 8 MB
  __bf16* kb    = (__bf16*)(ws + 8  * MB);  // 8 MB
  __bf16* vtb   = (__bf16*)(ws + 16 * MB);  // V transposed [B,H,dk,S], 8 MB
  __bf16* ctxt  = (__bf16*)(ws + 24 * MB);  // ctx transposed [B,H,dk,S], 8 MB
  __bf16* fracb = (__bf16*)(ws + 32 * MB);  // [B*S, D] bf16, 8 MB
  __bf16* xb    = (__bf16*)(ws + 40 * MB);  // 8 MB
  __bf16* wqb   = (__bf16*)(ws + 48 * MB);  // 2 MB each
  __bf16* wkb   = (__bf16*)(ws + 50 * MB);
  __bf16* wvb   = (__bf16*)(ws + 52 * MB);
  __bf16* wob   = (__bf16*)(ws + 54 * MB);
  float*  wc    = (float*) (ws + 56 * MB);  // 8 KB GL coefficients
  float*  outf  = (float*) (ws + 0  * MB);  // fp32 [4096,1024], aliases qb/kb

  const int NX = NROW * DIM;   // 4,194,304
  const int NW = DIM * DIM;    // 1,048,576
  cvt_bf16_kernel<<<(NX + 255) / 256, 256, 0, stream>>>(x,  xb,  NX);
  cvt_bf16_kernel<<<(NW + 255) / 256, 256, 0, stream>>>(Wq, wqb, NW);
  cvt_bf16_kernel<<<(NW + 255) / 256, 256, 0, stream>>>(Wk, wkb, NW);
  cvt_bf16_kernel<<<(NW + 255) / 256, 256, 0, stream>>>(Wv, wvb, NW);
  cvt_bf16_kernel<<<(NW + 255) / 256, 256, 0, stream>>>(Wo, wob, NW);
  glcoef_kernel<<<1, 32, 0, stream>>>(wc);

  // 256 m-tiles x 32 n-tiles(x32) = 8192 waves -> 2048 blocks, z = q/k/v
  proj_gemm_kernel<<<dim3(2048, 3), 128, 0, stream>>>(
      xb, wqb, wkb, wvb, bq, bk, bv, qb, kb, vtb);
  flash_attn_kernel<<<1024, 128, 0, stream>>>(qb, kb, vtb, ctxt);
  frac_toeplitz_kernel<<<1024, 128, 0, stream>>>(ctxt, wc, fracb);
  out_gemm_kernel<<<2048, 128, 0, stream>>>(fracb, wob, bo, outf);
  resid_ln_kernel<<<NROW, 256, 0, stream>>>(x, outf, lng, lnb, y);

  (void)in_sizes; (void)n_in; (void)out_size; (void)ws_size;
}
